// MockAWQQwen3_5Layer_39874476376664
// MI455X (gfx1250) — compile-verified, matrix operand order checked by reading, not run on hardware
//
#include <hip/hip_runtime.h>

// ---------------------------------------------------------------------------
// MI455X (gfx1250) AWQ transformer layer.
//  * int4 -> bf16 dequant into LDS, v_wmma_f32_16x16x32_bf16 (fp32 accum)
//  * A tiles staged with GLOBAL_LOAD_ASYNC_TO_LDS_B128 (ASYNCcnt path)
//  * dead k/v columns of the qkv GEMM skipped (only q is consumed)
// ---------------------------------------------------------------------------

typedef __attribute__((ext_vector_type(16))) __bf16 v16bf;
typedef __attribute__((ext_vector_type(8)))  __bf16 v8bf;
typedef __attribute__((ext_vector_type(8)))  float  v8f;

#define TOKENS 4096          // 2 * 2048
#define HDIM   2048
#define IDIM   8192

#define BM 128
#define BN 128
#define BK 64
#define SAK (BK + 8)         // LDS row stride (bf16 elems), keeps 16B alignment
#define SBK (BK + 8)

// -------------------------------- RMSNorm ----------------------------------
__global__ __launch_bounds__(256) void rmsnorm_bf16_kernel(
    const float* __restrict__ x, const float* __restrict__ w,
    __bf16* __restrict__ out)
{
    const int H = HDIM;
    const int row = blockIdx.x;
    const int tid = threadIdx.x;
    const float* xr = x + (size_t)row * H;

    float ss = 0.f;
    for (int i = tid; i < H; i += 256) { float v = xr[i]; ss += v * v; }

    __shared__ float red[256];
    red[tid] = ss;
    __syncthreads();
    for (int off = 128; off > 0; off >>= 1) {
        if (tid < off) red[tid] += red[tid + off];
        __syncthreads();
    }
    const float scale = rsqrtf(red[0] / (float)H + 1e-6f);

    __bf16* orow = out + (size_t)row * H;
    for (int i = tid; i < H; i += 256)
        orow[i] = (__bf16)(xr[i] * w[i] * scale);
}

// ------------------------- SiLU(gate) * up (bf16) --------------------------
// Elementwise i->i: safe to run in place over the gate buffer.
__global__ __launch_bounds__(256) void silu_mul_kernel(
    const __bf16* g, const __bf16* u, __bf16* o, long n)
{
    long i = (long)blockIdx.x * blockDim.x + threadIdx.x;
    const long stride = (long)gridDim.x * blockDim.x;
    for (; i < n; i += stride) {
        float fg = (float)g[i];
        float fu = (float)u[i];
        float s  = fg / (1.f + __expf(-fg));
        o[i] = (__bf16)(s * fu);
    }
}

// ----------------------- fused AWQ dequant + WMMA GEMM ---------------------
// C[M x Nout] = A[M x K](bf16) * dequant(W)[K x Nout], taking weight columns
// nOffset..nOffset+Nout of the Nfull-wide AWQ matrix.
// MODE 0: store bf16.  MODE 1: fp32 out = residual + acc.
__device__ __forceinline__ v16bf load_frag(const __bf16* rowp, int kk, int half)
{
    // 16-bit A/B layout: lane half 0 holds K = kk + {0..7, 16..23},
    //                    lane half 1 holds K = kk + {8..15, 24..31}.
    const v8bf lo = *(const v8bf*)(rowp + kk + half * 8);
    const v8bf hi = *(const v8bf*)(rowp + kk + 16 + half * 8);
    return __builtin_shufflevector(lo, hi, 0,1,2,3,4,5,6,7,8,9,10,11,12,13,14,15);
}

template <int MODE>
__global__ __launch_bounds__(256) void awq_gemm_kernel(
    const __bf16* __restrict__ A,    // [M][K], lda = K
    const int*    __restrict__ qw,   // [K][Nfull/8] packed int4
    const int*    __restrict__ qz,   // [K/128][Nfull/8] packed int4 zeros
    const float*  __restrict__ sc,   // [K/128][Nfull] scales
    const float*  __restrict__ res,  // residual (MODE 1), ldc = Nout
    void*         __restrict__ outp, // bf16 (MODE 0) or float (MODE 1)
    int K, int Nfull, int nOffset, int Nout)
{
    __shared__ __bf16 sA[BM * SAK];
    __shared__ __bf16 sB[BN * SBK];

    const int tid    = threadIdx.x;
    const int lane   = tid & 31;
    const int wv     = tid >> 5;       // 0..7
    const int warp_m = wv >> 1;        // 0..3  -> 32-row strip
    const int warp_n = wv & 1;         // 0..1  -> 64-col strip
    const int half   = lane >> 4;
    const int idx    = lane & 15;

    const int mtBase = blockIdx.x * BM;
    const int ntBase = blockIdx.y * BN;
    const int ldq    = Nfull >> 3;
    const int nColBase = nOffset + ntBase;   // weight-matrix column base
    const int wordBase = nColBase >> 3;

    v8f c[2][4] = {};

    const __bf16* aRow[2] = {
        &sA[(warp_m * 32 +  0 + idx) * SAK],
        &sA[(warp_m * 32 + 16 + idx) * SAK]
    };
    const __bf16* bRow[4] = {
        &sB[(warp_n * 64 +  0 + idx) * SBK],
        &sB[(warp_n * 64 + 16 + idx) * SBK],
        &sB[(warp_n * 64 + 32 + idx) * SBK],
        &sB[(warp_n * 64 + 48 + idx) * SBK]
    };

    for (int kt = 0; kt < K; kt += BK) {
        // ---- A tile: 128x64 bf16 via async global->LDS copies (ASYNCcnt) ----
        #pragma unroll
        for (int j = 0; j < 4; ++j) {
            const int vi  = tid + j * 256;   // 0..1023
            const int row = vi >> 3;         // 0..127
            const int c8  = vi & 7;          // 8-elem chunk
            const __bf16* gp = A + (size_t)(mtBase + row) * K + kt + c8 * 8;
            const unsigned ldsoff =
                (unsigned)(size_t)(&sA[row * SAK + c8 * 8]);
            asm volatile("global_load_async_to_lds_b128 %0, %1, off"
                         :: "v"(ldsoff), "v"(gp) : "memory");
        }

        // ---- dequant B tile while the async copies are in flight ----
        // One qw word = 8 columns at one k. AWQ nibble order: nibble i maps
        // to column (i<4 ? 2i : 2(i-4)+1) within the 8-col group.
        const int g = kt >> 7;               // AWQ group (BK=64 divides 128)
        #pragma unroll
        for (int it = 0; it < 4; ++it) {
            const int item = tid + it * 256;       // 0..1023
            const int k    = item >> 4;            // 0..63
            const int w8   = item & 15;            // word within tile row
            const int n0   = w8 * 8;               // local col base
            const unsigned wq = (unsigned)qw[(size_t)(kt + k) * ldq + wordBase + w8];
            const unsigned zq = (unsigned)qz[(size_t)g * ldq + wordBase + w8];
            const float4 s0 = *(const float4*)(sc + (size_t)g * Nfull + nColBase + n0);
            const float4 s1 = *(const float4*)(sc + (size_t)g * Nfull + nColBase + n0 + 4);
            const float sv[8] = {s0.x, s0.y, s0.z, s0.w, s1.x, s1.y, s1.z, s1.w};
            if (kt + BK < K)   // prefetch next k-tile's word (global_prefetch)
                __builtin_prefetch(qw + (size_t)(kt + BK + k) * ldq + wordBase + w8, 0, 0);
            #pragma unroll
            for (int i = 0; i < 8; ++i) {
                const int nloc = (i < 4) ? (2 * i) : (2 * (i - 4) + 1);
                const int wv_  = (int)((wq >> (4 * i)) & 15u);
                const int zv   = (int)((zq >> (4 * i)) & 15u);
                sB[(n0 + nloc) * SBK + k] = (__bf16)((float)(wv_ - zv) * sv[nloc]);
            }
        }

        asm volatile("s_wait_asynccnt 0x0" ::: "memory");
        __syncthreads();

        // ---- 2 WMMA k-steps of 32 over the 64-deep tile ----
        #pragma unroll
        for (int kk = 0; kk < BK; kk += 32) {
            v16bf aF[2], bF[4];
            #pragma unroll
            for (int t = 0; t < 2; ++t) aF[t] = load_frag(aRow[t], kk, half);
            #pragma unroll
            for (int t = 0; t < 4; ++t) bF[t] = load_frag(bRow[t], kk, half);
            #pragma unroll
            for (int ti = 0; ti < 2; ++ti)
                #pragma unroll
                for (int tj = 0; tj < 4; ++tj)
                    c[ti][tj] = __builtin_amdgcn_wmma_f32_16x16x32_bf16(
                        false, aF[ti], false, bF[tj],
                        (short)0, c[ti][tj], false, false);
        }
        __syncthreads();
    }

    // ---- epilogue: C/D layout -> (m = half*8 + r, n = idx) per 16x16 tile ----
    const int mBase = mtBase + warp_m * 32;
    const int nBase = ntBase + warp_n * 64;
    #pragma unroll
    for (int ti = 0; ti < 2; ++ti) {
        #pragma unroll
        for (int tj = 0; tj < 4; ++tj) {
            #pragma unroll
            for (int r = 0; r < 8; ++r) {
                const int m  = mBase + ti * 16 + half * 8 + r;
                const int nc = nBase + tj * 16 + idx;
                const size_t o = (size_t)m * Nout + nc;
                const float v = c[ti][tj][r];
                if (MODE == 0) {
                    ((__bf16*)outp)[o] = (__bf16)v;
                } else {
                    ((float*)outp)[o] = res[o] + v;
                }
            }
        }
    }
}

// --------------------------------- launch ----------------------------------
extern "C" void kernel_launch(void* const* d_in, const int* in_sizes, int n_in,
                              void* d_out, int out_size, void* d_ws, size_t ws_size,
                              hipStream_t stream)
{
    const float* x      = (const float*)d_in[0];
    const float* ln1_w  = (const float*)d_in[1];
    const float* ln2_w  = (const float*)d_in[2];
    const int*   qkv_qw = (const int*)d_in[3];
    const int*   qkv_qz = (const int*)d_in[4];
    const float* qkv_sc = (const float*)d_in[5];
    const int*   o_qw   = (const int*)d_in[6];
    const int*   o_qz   = (const int*)d_in[7];
    const float* o_sc   = (const float*)d_in[8];
    const int*   gu_qw  = (const int*)d_in[9];
    const int*   gu_qz  = (const int*)d_in[10];
    const float* gu_sc  = (const float*)d_in[11];
    const int*   dn_qw  = (const int*)d_in[12];
    const int*   dn_qz  = (const int*)d_in[13];
    const float* dn_sc  = (const float*)d_in[14];

    char* ws = (char*)d_ws;
    const size_t SZ_H  = (size_t)TOKENS * HDIM;   // 8M elems
    const size_t SZ_I  = (size_t)TOKENS * IDIM;   // 32M elems
    __bf16* h1   = (__bf16*)(ws);                                  // 16 MB
    __bf16* q    = (__bf16*)(ws + SZ_H * 2);                       // 16 MB
    float*  x2   = (float*) (ws + SZ_H * 4);                       // 32 MB
    __bf16* h2   = (__bf16*)(ws + SZ_H * 8);                       // 16 MB
    __bf16* gate = (__bf16*)(ws + SZ_H * 10);                      // 64 MB (reused as silu out)
    __bf16* up   = (__bf16*)(ws + SZ_H * 10 + SZ_I * 2);           // 64 MB
    float*  out  = (float*)d_out;

    const dim3 blk(256);

    // h1 = rmsnorm(x, ln1)
    rmsnorm_bf16_kernel<<<TOKENS, blk, 0, stream>>>(x, ln1_w, h1);

    // q = h1 @ W_qkv[:, :H]   (k/v columns are dead — skip them)
    awq_gemm_kernel<0><<<dim3(TOKENS / BM, HDIM / BN), blk, 0, stream>>>(
        h1, qkv_qw, qkv_qz, qkv_sc, nullptr, q, HDIM, 3 * HDIM, 0, HDIM);

    // x2 = x + q @ W_o
    awq_gemm_kernel<1><<<dim3(TOKENS / BM, HDIM / BN), blk, 0, stream>>>(
        q, o_qw, o_qz, o_sc, x, x2, HDIM, HDIM, 0, HDIM);

    // h2 = rmsnorm(x2, ln2)
    rmsnorm_bf16_kernel<<<TOKENS, blk, 0, stream>>>(x2, ln2_w, h2);

    // gate = h2 @ W_gu[:, :I] ; up = h2 @ W_gu[:, I:]
    awq_gemm_kernel<0><<<dim3(TOKENS / BM, IDIM / BN), blk, 0, stream>>>(
        h2, gu_qw, gu_qz, gu_sc, nullptr, gate, HDIM, 2 * IDIM, 0, IDIM);
    awq_gemm_kernel<0><<<dim3(TOKENS / BM, IDIM / BN), blk, 0, stream>>>(
        h2, gu_qw, gu_qz, gu_sc, nullptr, up, HDIM, 2 * IDIM, IDIM, IDIM);

    // gate = silu(gate) * up   (in place, elementwise)
    silu_mul_kernel<<<4096, blk, 0, stream>>>(gate, up, gate, (long)SZ_I);

    // out = x2 + mlp_in @ W_dn
    awq_gemm_kernel<1><<<dim3(TOKENS / BM, HDIM / BN), blk, 0, stream>>>(
        gate, dn_qw, dn_qz, dn_sc, x2, out, IDIM, HDIM, 0, HDIM);

    (void)in_sizes; (void)n_in; (void)out_size; (void)ws_size;
}
